// ShapeRetrieval_46282567582063
// MI455X (gfx1250) — matrix operand order repaired and checked by reading, not compiled, not run on hardware
//
#include <hip/hip_runtime.h>

typedef __bf16 bf16;
typedef __attribute__((ext_vector_type(16))) __bf16 v16bf;
typedef __attribute__((ext_vector_type(8)))  __bf16 v8bf;
typedef __attribute__((ext_vector_type(8)))  float  v8f;

#define DDIM   128
#define GCHUNK 4096
#define GTILE  32    // gallery rows per block iteration (2 x 16-col WMMA tiles)
#define QPB    128   // queries per block (grid.y picks which half)
#define TOPK   10

#define GBUF_BYTES  (2 * GTILE * DDIM * sizeof(bf16))            // 16384
#define SIMS_BYTES  (8 * 2 * 16 * 17 * sizeof(float))            // 17408
#define QSTG_BYTES  (QPB * DDIM * sizeof(bf16))                  // 32768
#define PHASE2      (GBUF_BYTES + SIMS_BYTES)                    // 33792
#define SMEM_BYTES  (PHASE2 > QSTG_BYTES ? PHASE2 : QSTG_BYTES)  // 33792

// ---------------- insertion into a register-resident sorted (desc) top-10 ----
__device__ __forceinline__ void topk_insert(float (&tv)[TOPK], int (&ti)[TOPK],
                                            float v, int gi) {
    if (v <= tv[TOPK - 1]) return;
    int pos = TOPK - 1;
#pragma unroll
    for (int k = TOPK - 1; k > 0; --k) {
        if (v > tv[k - 1]) {
            tv[k] = tv[k - 1];
            ti[k] = ti[k - 1];
            pos = k - 1;
        }
    }
    tv[pos] = v;
    ti[pos] = gi;
}

// ---------------- kernel 1: L2-normalize queries, emit bf16 -----------------
__global__ __launch_bounds__(DDIM) void qnorm_kernel(const float* __restrict__ q,
                                                     bf16* __restrict__ qn) {
    __shared__ float red[DDIM];
    const int r = blockIdx.x;
    const int t = threadIdx.x;
    float v = q[r * DDIM + t];
    red[t] = v * v;
    __syncthreads();
#pragma unroll
    for (int s = DDIM / 2; s > 0; s >>= 1) {
        if (t < s) red[t] += red[t + s];
        __syncthreads();
    }
    float sc = 1.0f / fmaxf(sqrtf(red[0]), 1e-12f);
    qn[r * DDIM + t] = (bf16)(v * sc);
}

// ---------------- kernel 2: fused normalize + bf16 WMMA sims + block top-10 -
__global__ __launch_bounds__(256) void sim_topk_kernel(
    const float* __restrict__ gal, const bf16* __restrict__ qn,
    float* __restrict__ pvals, int* __restrict__ pidx, int G, int Qtot) {
    // One raw LDS arena, reused across two barrier-separated phases:
    //   phase 1: 32 KB query staging (dead once A-fragments are in VGPRs)
    //   phase 2: [2][32][128] bf16 gallery double-buffer + per-wave sims spill
    __shared__ __align__(32) unsigned char smem[SMEM_BYTES];
    bf16 (*qs)[DDIM]          = (bf16(*)[DDIM])smem;
    bf16 (*gb)[GTILE][DDIM]   = (bf16(*)[GTILE][DDIM])smem;
    float (*sims)[2][16][17]  = (float(*)[2][16][17])(smem + GBUF_BYTES);

    const int t     = threadIdx.x;
    const int lane  = t & 31;
    const int wave  = t >> 5;
    const int qhalf = blockIdx.y;

    // ---- phase 1: stage 128 normalized bf16 query rows, hoist A to VGPRs ----
    {
        const uint4* src4 = (const uint4*)(qn + (size_t)qhalf * QPB * DDIM);
        uint4* dst4 = (uint4*)&qs[0][0];
#pragma unroll
        for (int i = 0; i < 8; ++i) dst4[t + i * 256] = src4[t + i * 256];
    }
    __syncthreads();

    // WMMA fragment addressing (contiguous k = 16h + s for both A and B)
    const int qrow = (wave << 4) + (lane & 15);   // A: M row = query
    const int ka   = (lane >> 4) << 3;            // unused: A loads use kb (K base 0/16)
    const int nrow = lane & 15;                    // B: N row within 16-col tile
    const int kb   = (lane >> 4) << 4;            // B: K base 0/16
    const int mb   = (lane >> 4) << 3;            // C: M base 0/8

    v16bf Af[4];
#pragma unroll
    for (int kk4 = 0; kk4 < 4; ++kk4) {
        const int kk = kk4 * 32;
        v8bf a0 = *(const v8bf*)&qs[qrow][kk + kb];
        v8bf a1 = *(const v8bf*)&qs[qrow][kk + kb + 8];
#pragma unroll
        for (int i = 0; i < 8; ++i) { Af[kk4][i] = a0[i]; Af[kk4][8 + i] = a1[i]; }
    }
    __syncthreads();  // queries read; arena may be reused as gallery/sims

    const int gstart = (int)blockIdx.x * GCHUNK;

    float tv[TOPK];
    int   ti[TOPK];
#pragma unroll
    for (int k = 0; k < TOPK; ++k) { tv[k] = -3.0e38f; ti[k] = 0; }

    const int r  = t >> 3;          // gallery row within 32-row tile (8 lanes/row)
    const int cb = (t & 7) * 16;    // 16-float column chunk per thread
    const int jt = lane >> 4;       // which 16-col tile this lane scans in top-k
    const int ql = lane & 15;       // query owned by this lane in top-k

    // ---- prologue: load + normalize tile 0 into buffer 0 ----
    {
        const int  grow = gstart + r;
        const bool inb  = grow < G;
        const float* gp = gal + (long)(inb ? grow : 0) * DDIM + cb;
        const float msk = inb ? 1.0f : 0.0f;
        float v16[16];
#pragma unroll
        for (int j = 0; j < 4; ++j) {
            float4 f = ((const float4*)gp)[j];
            v16[4 * j + 0] = f.x * msk; v16[4 * j + 1] = f.y * msk;
            v16[4 * j + 2] = f.z * msk; v16[4 * j + 3] = f.w * msk;
        }
        float ss = 0.0f;
#pragma unroll
        for (int i = 0; i < 16; ++i) ss += v16[i] * v16[i];
        ss += __shfl_xor(ss, 1, 32);
        ss += __shfl_xor(ss, 2, 32);
        ss += __shfl_xor(ss, 4, 32);
        const float sc = 1.0f / fmaxf(sqrtf(ss), 1e-12f);
#pragma unroll
        for (int i = 0; i < 16; ++i) gb[0][r][cb + i] = (bf16)(v16[i] * sc);
    }
    __syncthreads();

    // ---- main loop: 1 barrier/iteration, next-tile loads overlap WMMA ----
    for (int it = 0; it < GCHUNK / GTILE; ++it) {
        const int g0  = gstart + it * GTILE;
        const int cur = it & 1, nxt = cur ^ 1;

        // issue next tile's global loads NOW (consumed after the WMMA burst)
        const int  grow = g0 + GTILE + r;
        const bool inb  = grow < G;
        const float* gp = gal + (long)(inb ? grow : 0) * DDIM + cb;
        const float msk = inb ? 1.0f : 0.0f;
        float v16[16];
#pragma unroll
        for (int j = 0; j < 4; ++j) {
            float4 f = ((const float4*)gp)[j];
            v16[4 * j + 0] = f.x * msk; v16[4 * j + 1] = f.y * msk;
            v16[4 * j + 2] = f.z * msk; v16[4 * j + 3] = f.w * msk;
        }

        // --- 8-WMMA burst on the current buffer ---
        v16bf Bf[8];
#pragma unroll
        for (int j = 0; j < 2; ++j)
#pragma unroll
            for (int kk4 = 0; kk4 < 4; ++kk4)
                Bf[j * 4 + kk4] =
                    *(const v16bf*)&gb[cur][j * 16 + nrow][kk4 * 32 + kb];

        v8f c0 = {}, c1 = {};
#pragma unroll
        for (int kk4 = 0; kk4 < 4; ++kk4) {
            c0 = __builtin_amdgcn_wmma_f32_16x16x32_bf16(
                false, Af[kk4], false, Bf[kk4],     (short)0, c0, false, false);
            c1 = __builtin_amdgcn_wmma_f32_16x16x32_bf16(
                false, Af[kk4], false, Bf[4 + kk4], (short)0, c1, false, false);
        }

        // wave-private sims spill (LDS is in-order per wave: no block barrier)
#pragma unroll
        for (int i = 0; i < 8; ++i) {
            sims[wave][0][mb + i][nrow] = c0[i];
            sims[wave][1][mb + i][nrow] = c1[i];
        }

        // --- normalize + store next tile into the other buffer ---
        {
            float ss = 0.0f;
#pragma unroll
            for (int i = 0; i < 16; ++i) ss += v16[i] * v16[i];
            ss += __shfl_xor(ss, 1, 32);
            ss += __shfl_xor(ss, 2, 32);
            ss += __shfl_xor(ss, 4, 32);
            const float sc = 1.0f / fmaxf(sqrtf(ss), 1e-12f);
#pragma unroll
            for (int i = 0; i < 16; ++i) gb[nxt][r][cb + i] = (bf16)(v16[i] * sc);
        }

        // --- top-k: all 32 lanes active; lane owns (query ql, tile jt) ---
#pragma unroll
        for (int n = 0; n < 16; ++n) {
            const int gi = g0 + jt * 16 + n;
            float v = sims[wave][jt][ql][n];
            v = (gi < G) ? v : -3.0e38f;
            topk_insert(tv, ti, v, gi);
        }

        __syncthreads();  // next buffer ready; current buffer reads complete
    }

    // two partial lists per query per block (one per half-wave)
    {
        const int  qglob = qhalf * QPB + (wave << 4) + ql;
        const int  slot  = (int)blockIdx.x * 2 + jt;
        const long base  = ((long)slot * Qtot + qglob) * TOPK;
#pragma unroll
        for (int k = 0; k < TOPK; ++k) {
            pvals[base + k] = tv[k];
            pidx[base + k]  = ti[k];
        }
    }
}

// ---------------- kernel 3: merge per-block partial top-10s -----------------
__global__ __launch_bounds__(64) void merge_topk_kernel(
    const float* __restrict__ pvals, const int* __restrict__ pidx,
    const int* __restrict__ shape_idx, float* __restrict__ out,
    int NBP, int Qtot) {
    __shared__ float lv[64 * TOPK];
    __shared__ int   li[64 * TOPK];
    const int q = blockIdx.x;
    const int t = threadIdx.x;

    float tv[TOPK];
    int   ti[TOPK];
#pragma unroll
    for (int k = 0; k < TOPK; ++k) { tv[k] = -3.0e38f; ti[k] = 0; }

    const int total = NBP * TOPK;
    for (int c = t; c < total; c += 64) {
        const int b = c / TOPK, k = c % TOPK;
        const long off = ((long)b * Qtot + q) * TOPK + k;
        topk_insert(tv, ti, pvals[off], pidx[off]);
    }
#pragma unroll
    for (int k = 0; k < TOPK; ++k) {
        lv[t * TOPK + k] = tv[k];
        li[t * TOPK + k] = ti[k];
    }
    __syncthreads();
    if (t == 0) {
        for (int k = 0; k < TOPK; ++k) {
            float best = -3.0e38f;
            int bj = 0;
            for (int j = 0; j < 64 * TOPK; ++j)
                if (lv[j] > best) { best = lv[j]; bj = j; }
            const int gi = li[bj];
            lv[bj] = -3.0e38f;
            out[q * TOPK + k]                   = best;                  // vals
            out[Qtot * TOPK + q * TOPK + k]     = (float)shape_idx[gi];  // retrieved
            out[2 * Qtot * TOPK + q * TOPK + k] = (float)gi;             // idx
        }
    }
}

// ---------------- launch ----------------------------------------------------
extern "C" void kernel_launch(void* const* d_in, const int* in_sizes, int n_in,
                              void* d_out, int out_size, void* d_ws, size_t ws_size,
                              hipStream_t stream) {
    const float* q         = (const float*)d_in[0];
    const float* gal       = (const float*)d_in[1];
    const int*   shape_idx = (const int*)d_in[2];
    // d_in[3] is K on device; reference setup fixes K=10 (TOPK).

    const int Q   = in_sizes[0] / DDIM;             // 256
    const int G   = in_sizes[1] / DDIM;             // 1,000,000
    const int NB  = (G + GCHUNK - 1) / GCHUNK;      // 245
    const int NBP = NB * 2;                         // 2 partial lists per block

    // workspace layout: [qn bf16][partial vals f32][partial idx i32]
    bf16* qn = (bf16*)d_ws;
    size_t off = (((size_t)Q * DDIM * sizeof(bf16)) + 255) & ~(size_t)255;
    float* pvals = (float*)((char*)d_ws + off);
    int*   pidx  = (int*)((char*)d_ws + off + (size_t)NBP * Q * TOPK * sizeof(float));

    qnorm_kernel<<<Q, DDIM, 0, stream>>>(q, qn);

    dim3 grid(NB, Q / QPB);
    sim_topk_kernel<<<grid, 256, 0, stream>>>(gal, qn, pvals, pidx, G, Q);

    merge_topk_kernel<<<Q, 64, 0, stream>>>(pvals, pidx, shape_idx,
                                            (float*)d_out, NBP, Q);
}
